// WindowAttention_46651934769823
// MI455X (gfx1250) — matrix operand order repaired and checked by reading, not compiled, not run on hardware
//
#include <hip/hip_runtime.h>
#include <math.h>

// ---------------------------------------------------------------------------
// Sliding-window attention (window=127) for shape (8, 4094, 8, 64), fp32 io.
// Strategy: per-block flash-style tile, f16 WMMA (v_wmma_f32_16x16x32_f16)
// with fp32 softmax. One workgroup = one (batch, head, 128-query chunk).
// ---------------------------------------------------------------------------

typedef __attribute__((ext_vector_type(16))) _Float16 v16h;
typedef __attribute__((ext_vector_type(8)))  _Float16 v8h;
typedef __attribute__((ext_vector_type(4)))  _Float16 v4h;
typedef __attribute__((ext_vector_type(8)))  float    v8f;
typedef __attribute__((ext_vector_type(4)))  float    v4f;

#define S_LEN  4094
#define NHEAD  8
#define HDIM   64
#define QC     128       // queries per chunk
#define KC     256       // key window slots (254 real, padded to 256)
#define NCHUNK 32        // ceil(4094/128)
#define W_RAD  63        // (127-1)/2
#define KSTR   72        // LDS stride (halfs) for K rows   (bank-conflict pad)
#define VSTR   264       // LDS stride (halfs) for V^T rows (bank-conflict pad)
#define PSTR   264       // LDS stride (halfs) for P rows

__device__ __forceinline__ v16h cat8(v8h lo, v8h hi) {
    return __builtin_shufflevector(lo, hi, 0,1,2,3,4,5,6,7,8,9,10,11,12,13,14,15);
}

__global__ void __launch_bounds__(256, 1)
window_attn_kernel(const float* __restrict__ q,
                   const float* __restrict__ k,
                   const float* __restrict__ v,
                   float* __restrict__ out)
{
    __shared__ _Float16 Klds[KC * KSTR];          // [key][dim]   36864 B
    __shared__ _Float16 Vtlds[HDIM * VSTR];       // [dim][key]   33792 B
    __shared__ _Float16 Plds[8 * 16 * PSTR];      // per-wave P   67584 B

    const int blk  = blockIdx.x;
    const int c    = blk & (NCHUNK - 1);
    const int n    = (blk >> 5) & (NHEAD - 1);
    const int b    = blk >> 8;

    const int tid    = threadIdx.x;
    const int wave   = tid >> 5;
    const int lane   = tid & 31;
    const int lhalf  = lane >> 4;      // 0: lanes 0-15, 1: lanes 16-31
    const int lane15 = lane & 15;

    // ---------------- Phase 1: stage K (row-major f16) and V^T into LDS ----
    {
        const int j    = tid;                       // key slot 0..255
        const int kpos = c * QC - W_RAD + j;        // global key position
        if (kpos >= 0 && kpos < S_LEN && j < KC) {
            const float* kr = k + ((b * S_LEN + kpos) * NHEAD + n) * HDIM;
            const float* vr = v + ((b * S_LEN + kpos) * NHEAD + n) * HDIM;
            #pragma unroll
            for (int d = 0; d < HDIM; d += 4) {
                v4f kv = *(const v4f*)(kr + d);
                v4f vv = *(const v4f*)(vr + d);
                v4h kh = { (_Float16)kv.x, (_Float16)kv.y,
                           (_Float16)kv.z, (_Float16)kv.w };
                *(v4h*)(&Klds[j * KSTR + d]) = kh;
                Vtlds[(d + 0) * VSTR + j] = (_Float16)vv.x;
                Vtlds[(d + 1) * VSTR + j] = (_Float16)vv.y;
                Vtlds[(d + 2) * VSTR + j] = (_Float16)vv.z;
                Vtlds[(d + 3) * VSTR + j] = (_Float16)vv.w;
            }
        } else {
            #pragma unroll
            for (int d = 0; d < HDIM; d += 4) {
                v4h zz = { (_Float16)0.f, (_Float16)0.f,
                           (_Float16)0.f, (_Float16)0.f };
                *(v4h*)(&Klds[j * KSTR + d]) = zz;
                Vtlds[(d + 0) * VSTR + j] = (_Float16)0.f;
                Vtlds[(d + 1) * VSTR + j] = (_Float16)0.f;
                Vtlds[(d + 2) * VSTR + j] = (_Float16)0.f;
                Vtlds[(d + 3) * VSTR + j] = (_Float16)0.f;
            }
        }
    }
    __syncthreads();

    // ---------------- Q A-fragments (16x32 f16 layout, 2 k-slices) ---------
    // A 16x32 layout: lanes 0-15 row M=lane, K chunks {0..7, 16..23};
    // lanes 16-31 row M=lane-16, K chunks {8..15, 24..31}.
    v16h qa[2];
    {
        const int qrow = wave * 16 + lane15;
        int qpos = c * QC + qrow;
        if (qpos >= S_LEN) qpos = S_LEN - 1;       // padded rows: never stored
        const float* qr = q + ((b * S_LEN + qpos) * NHEAD + n) * HDIM;
        const float sc = 0.125f;                   // 64^-0.5
        #pragma unroll
        for (int kt = 0; kt < 2; ++kt) {
            const int base = kt * 32 + lhalf * 8;
            v4f a0 = *(const v4f*)(qr + base + 0);
            v4f a1 = *(const v4f*)(qr + base + 4);
            v4f a2 = *(const v4f*)(qr + base + 16);
            v4f a3 = *(const v4f*)(qr + base + 20);
            v16h f;
            f[0]=(_Float16)(a0.x*sc); f[1]=(_Float16)(a0.y*sc);
            f[2]=(_Float16)(a0.z*sc); f[3]=(_Float16)(a0.w*sc);
            f[4]=(_Float16)(a1.x*sc); f[5]=(_Float16)(a1.y*sc);
            f[6]=(_Float16)(a1.z*sc); f[7]=(_Float16)(a1.w*sc);
            f[8]=(_Float16)(a2.x*sc); f[9]=(_Float16)(a2.y*sc);
            f[10]=(_Float16)(a2.z*sc); f[11]=(_Float16)(a2.w*sc);
            f[12]=(_Float16)(a3.x*sc); f[13]=(_Float16)(a3.y*sc);
            f[14]=(_Float16)(a3.z*sc); f[15]=(_Float16)(a3.w*sc);
            qa[kt] = f;
        }
    }

    // ---------------- GEMM 1: S = Q * K^T  (16x256 per wave) ---------------
    v8f acc[16];
    #pragma unroll
    for (int nt = 0; nt < 16; ++nt) {
        v8f a = {};
        #pragma unroll
        for (int kt = 0; kt < 2; ++kt) {
            // B 32x16 layout: lane col N=lane&15; lanes 0-15 K={0..7,16..23},
            // lanes 16-31 K={8..15,24..31} (matches A fragment order)
            const int key = nt * 16 + lane15;
            const _Float16* src = &Klds[key * KSTR + kt * 32 + lhalf * 8];
            v8h blo = *(const v8h*)(src);
            v8h bhi = *(const v8h*)(src + 16);
            a = __builtin_amdgcn_wmma_f32_16x16x32_f16(
                    false, qa[kt], false, cat8(blo, bhi),
                    (short)0, a, false, false);
        }
        acc[nt] = a;
    }

    // ---------------- Mask + softmax (fp32) --------------------------------
    // C/D layout: row m = r + 8*lhalf, col = lane15 within each 16-col tile.
    float rmax[8], rsum[8];
    #pragma unroll
    for (int r = 0; r < 8; ++r) rmax[r] = -3.0e38f;

    #pragma unroll
    for (int nt = 0; nt < 16; ++nt) {
        const int j    = nt * 16 + lane15;           // window column 0..255
        const int kpos = c * QC - W_RAD + j;
        const bool kvalid = (kpos >= 0) && (kpos < S_LEN) && (j < 254);
        #pragma unroll
        for (int r = 0; r < 8; ++r) {
            const int i = wave * 16 + r + 8 * lhalf; // query row in chunk
            const bool ok = kvalid && (j >= i) && (j <= i + 2 * W_RAD);
            float sv = ok ? acc[nt][r] : -3.0e38f;
            acc[nt][r] = sv;
            rmax[r] = fmaxf(rmax[r], sv);
        }
    }
    #pragma unroll
    for (int m = 1; m < 16; m <<= 1) {
        #pragma unroll
        for (int r = 0; r < 8; ++r)
            rmax[r] = fmaxf(rmax[r], __shfl_xor(rmax[r], m, 32));
    }
    #pragma unroll
    for (int r = 0; r < 8; ++r) rsum[r] = 0.f;
    #pragma unroll
    for (int nt = 0; nt < 16; ++nt) {
        #pragma unroll
        for (int r = 0; r < 8; ++r) {
            float e = __expf(acc[nt][r] - rmax[r]);  // masked -> exp(-big)=0
            acc[nt][r] = e;
            rsum[r] += e;
        }
    }
    #pragma unroll
    for (int m = 1; m < 16; m <<= 1) {
        #pragma unroll
        for (int r = 0; r < 8; ++r)
            rsum[r] += __shfl_xor(rsum[r], m, 32);
    }

    // ---------------- P (unnormalized, f16) to per-wave LDS tile -----------
    _Float16* pw = &Plds[wave * 16 * PSTR];
    #pragma unroll
    for (int nt = 0; nt < 16; ++nt) {
        #pragma unroll
        for (int r = 0; r < 8; ++r) {
            pw[(r + 8 * lhalf) * PSTR + nt * 16 + lane15] =
                (_Float16)acc[nt][r];
        }
    }

    // ---------------- GEMM 2: Z = P * V  (16x64 per wave) ------------------
    v8f oacc[4];
    #pragma unroll
    for (int dt = 0; dt < 4; ++dt) { v8f z = {}; oacc[dt] = z; }

    #pragma unroll
    for (int kt2 = 0; kt2 < 8; ++kt2) {
        // A fragment of P: row = lane15, K chunks {base..base+7, base+16..+23}
        const int abase = kt2 * 32 + lhalf * 8;
        const _Float16* ap = &pw[lane15 * PSTR + abase];
        v8h alo = *(const v8h*)(ap);
        v8h ahi = *(const v8h*)(ap + 16);
        const v16h afrag = cat8(alo, ahi);
        #pragma unroll
        for (int dt = 0; dt < 4; ++dt) {
            // B fragment of V: col N = dim = dt*16 + lane15, K = keys
            const int dim = dt * 16 + lane15;
            const _Float16* bp = &Vtlds[dim * VSTR + kt2 * 32 + lhalf * 8];
            v8h blo = *(const v8h*)(bp);
            v8h bhi = *(const v8h*)(bp + 16);
            oacc[dt] = __builtin_amdgcn_wmma_f32_16x16x32_f16(
                           false, afrag, false, cat8(blo, bhi),
                           (short)0, oacc[dt], false, false);
        }
    }

    // ---------------- Normalize + store ------------------------------------
    #pragma unroll
    for (int r = 0; r < 8; ++r) {
        const int m    = r + 8 * lhalf;
        const int qpos = c * QC + wave * 16 + m;
        if (qpos < S_LEN) {
            const float inv = 1.0f / rsum[r];
            float* orow = out + ((b * S_LEN + qpos) * NHEAD + n) * HDIM;
            #pragma unroll
            for (int dt = 0; dt < 4; ++dt)
                orow[dt * 16 + lane15] = oacc[dt][r] * inv;
        }
    }
}

extern "C" void kernel_launch(void* const* d_in, const int* in_sizes, int n_in,
                              void* d_out, int out_size, void* d_ws, size_t ws_size,
                              hipStream_t stream) {
    const float* q = (const float*)d_in[0];
    const float* k = (const float*)d_in[1];
    const float* v = (const float*)d_in[2];
    float* out = (float*)d_out;

    const int nblocks = 8 /*batch*/ * NHEAD * NCHUNK;   // 2048
    window_attn_kernel<<<dim3(nblocks), dim3(256), 0, stream>>>(q, k, v, out);
}
